// BidirTLSTMLayer_45226005627006
// MI455X (gfx1250) — matrix operand order repaired and checked by reading, not compiled, not run on hardware
//
#include <hip/hip_runtime.h>
#include <cstddef>

typedef __attribute__((ext_vector_type(16))) _Float16 v16h;
typedef __attribute__((ext_vector_type(8)))  float    v8f;
typedef __attribute__((ext_vector_type(4)))  float    f32x4;

#if __has_builtin(__builtin_amdgcn_s_barrier_signal)
#define BAR_SIGNAL() __builtin_amdgcn_s_barrier_signal(-1)
#else
#define BAR_SIGNAL() asm volatile("s_barrier_signal -1" ::: "memory")
#endif
#if __has_builtin(__builtin_amdgcn_s_barrier_wait)
#define BAR_WAIT() __builtin_amdgcn_s_barrier_wait(-1)
#else
#define BAR_WAIT() asm volatile("s_barrier_wait -1" ::: "memory")
#endif
#define WAIT_DS0() asm volatile("s_wait_dscnt 0x0" ::: "memory")

namespace {

constexpr int kS = 2048;   // sequence length
constexpr int kB = 64;     // batch
constexpr int kI = 128;    // input dim
constexpr int kH = 128;    // hidden dim
constexpr int kThreads = 512;   // 16 wave32s per block, one block per direction

// LDS layout in _Float16 units. One "frag" = 32 lanes x 16 halves = 512 halves = 1KB,
// stored fragment-linear: lane l's 16 halves live at frag_base + l*16 (32B contiguous).
constexpr int kFragH  = 512;
constexpr int kWcbOff = 0;                       // 256 frags: gate B ([W_ih;W_hh]^T), K=256
constexpr int kWdbOff = kWcbOff + 256 * kFragH;  // 32 frags:  W_d^T B-fragments
constexpr int kHaOff  = kWdbOff + 32 * kFragH;   // 16 frags:  h A-fragments (f16)
constexpr int kCaOff  = kHaOff  + 16 * kFragH;   // 16 frags:  c A-fragments (f16)
constexpr int kLdsHalfs = kCaOff + 16 * kFragH;  // 163840 halves = 327680 B = 320 KB

struct Params {
  const float* x;        // [S,B,I]
  const float* h0;       // [2,B,H]
  const float* c0;       // [2,B,H]
  const float* dt;       // [B,S]
  const float* W_ih[2];  // [4H,I]
  const float* W_hh[2];  // [4H,H]
  const float* b_ih[2];  // [4H]
  const float* b_hh[2];  // [4H]
  const float* W_d[2];   // [H,H]
  const float* b_d[2];   // [H]
  float* out;            // [S,B,2H]
};

__device__ __forceinline__ v8f wmma16(v16h a, v16h b, v8f c) {
  // D = A(16x32 f16) x B(32x16 f16) + C(16x16 f32)
  return __builtin_amdgcn_wmma_f32_16x16x32_f16(false, a, false, b, (short)0, c,
                                                false, false);
}

__device__ __forceinline__ float fastrcp(float x) { return __builtin_amdgcn_rcpf(x); }

#if __has_builtin(__builtin_amdgcn_tanhf)
__device__ __forceinline__ float tanh_fast(float x) { return __builtin_amdgcn_tanhf(x); }
__device__ __forceinline__ float sigm(float x) {
  return __builtin_amdgcn_tanhf(0.5f * x) * 0.5f + 0.5f;
}
#else
__device__ __forceinline__ float tanh_fast(float x) {
  return 1.0f - 2.0f * fastrcp(__expf(2.0f * x) + 1.0f);
}
__device__ __forceinline__ float sigm(float x) {
  return fastrcp(1.0f + __expf(-x));
}
#endif

// Prepare the x A-fragments and time-decay broadcast for step index s;
// prefetch step sp. Pure global/VALU/permute work: runs while the
// "writes-done" barrier round completes in the background.
__device__ __forceinline__ void prep_step(const Params& p, int s, int sp,
                                          int bt, int dk, int hi,
                                          v16h ax[4], float wr8[8]) {
  const float dtv = p.dt[(size_t)(bt * 16 + dk) * kS + s];
  const float wsc = fastrcp(__logf(2.718281828459045f + dtv));
#pragma unroll
  for (int r = 0; r < 8; ++r) wr8[r] = __shfl(wsc, r + 8 * hi, 32);

  const float* xrow = p.x + ((size_t)s * kB + bt * 16 + dk) * kI + hi * 8;
#pragma unroll
  for (int kf = 0; kf < 4; ++kf) {
    const float* px = xrow + kf * 32;
    f32x4 r0 = *(const f32x4*)(px);
    f32x4 r1 = *(const f32x4*)(px + 4);
    f32x4 r2 = *(const f32x4*)(px + 16);
    f32x4 r3 = *(const f32x4*)(px + 20);
    v16h a;
#pragma unroll
    for (int t = 0; t < 4; ++t) {
      a[t]      = (_Float16)r0[t];
      a[4 + t]  = (_Float16)r1[t];
      a[8 + t]  = (_Float16)r2[t];
      a[12 + t] = (_Float16)r3[t];
    }
    ax[kf] = a;
  }
  __builtin_prefetch(p.x + ((size_t)sp * kB + bt * 16 + dk) * kI, 0, 1);
}

} // namespace

__global__ __launch_bounds__(kThreads, 1)
void tlstm_bidir_kernel(Params p) {
  __shared__ alignas(32) _Float16 lds[kLdsHalfs];

  const int dir  = blockIdx.x;      // 0 = forward, 1 = reverse
  const int tid  = threadIdx.x;
  const int lane = tid & 31;
  const int wave = tid >> 5;
  const int dk   = lane & 15;       // column-within-tile for B/C layouts
  const int hi   = lane >> 4;       // half-wave id
  const int bt   = wave & 3;        // batch tile (16 rows of B=64)
  const int j0   = wave >> 2;       // wave owns hidden col-tiles j0 and j0+4

  const float* Wih = p.W_ih[dir];
  const float* Whh = p.W_hh[dir];
  const float* Wd  = p.W_d[dir];

  // ------------------------------------------------------------------
  // One-time staging of weights / initial state into fragment-linear LDS.
  // ------------------------------------------------------------------
  for (int slot = tid; slot < 256 * 32; slot += kThreads) {
    const int F = slot >> 5, l = slot & 31;
    const int kf = F & 7, g = (F >> 3) & 3, j = F >> 5;
    const int n  = g * 128 + j * 16 + (l & 15);
    const int kk = kf * 32 + ((l >> 4) << 3);  // K runs [kk..kk+7], [kk+16..kk+23]
    const float* src = (kk < 128) ? (Wih + n * kI + kk) : (Whh + n * kH + (kk - 128));
    _Float16* dst = &lds[kWcbOff + F * kFragH + l * 16];
#pragma unroll
    for (int t = 0; t < 16; ++t) dst[t] = (_Float16)src[t + (t & 8)];
  }
  for (int slot = tid; slot < 32 * 32; slot += kThreads) {
    const int F = slot >> 5, l = slot & 31;
    const int kf = F & 3, j = F >> 2;
    const int n  = j * 16 + (l & 15);
    const int kk = kf * 32 + ((l >> 4) << 3);
    const float* src = Wd + n * kH + kk;
    _Float16* dst = &lds[kWdbOff + F * kFragH + l * 16];
#pragma unroll
    for (int t = 0; t < 16; ++t) dst[t] = (_Float16)src[t + (t & 8)];
  }
  // h0/c0 A-frags. A (16-bit 16x32) layout: lanes 0-15 hold M=lane, K runs
  // [k0..k0+7] and [k0+16..k0+23]; lanes 16-31 hold runs shifted by +8.
  for (int slot = tid; slot < 16 * 32; slot += kThreads) {
    const int F = slot >> 5, l = slot & 31;
    const int kf = F & 3, btf = F >> 2;
    const int m  = btf * 16 + (l & 15);
    const int k0 = kf * 32 + ((l >> 4) << 3);
    const float* hs = p.h0 + (size_t)(dir * kB + m) * kH;
    const float* cs = p.c0 + (size_t)(dir * kB + m) * kH;
    _Float16* hd = &lds[kHaOff + F * kFragH + l * 16];
    _Float16* cd = &lds[kCaOff + F * kFragH + l * 16];
#pragma unroll
    for (int t = 0; t < 8; ++t) {
      hd[t]     = (_Float16)hs[k0 + t];
      hd[8 + t] = (_Float16)hs[k0 + 16 + t];
      cd[t]     = (_Float16)cs[k0 + t];
      cd[8 + t] = (_Float16)cs[k0 + 16 + t];
    }
  }

  // Per-wave persistent state: biases (per column) + f32 cell patch.
  float bias[2][4];
  float bdv[2];
  v8f creg[2];
#pragma unroll
  for (int u = 0; u < 2; ++u) {
    const int j   = j0 + u * 4;
    const int col = j * 16 + dk;
#pragma unroll
    for (int g = 0; g < 4; ++g) {
      const int n = g * 128 + col;
      bias[u][g] = p.b_ih[dir][n] + p.b_hh[dir][n];
    }
    bdv[u] = p.b_d[dir][col];
#pragma unroll
    for (int r = 0; r < 8; ++r)
      creg[u][r] = p.c0[(size_t)(dir * kB + bt * 16 + r + 8 * hi) * kH + col];
  }
  __syncthreads();   // staging visible to whole workgroup

  const v8f vzero = {};

  // Software pipeline prologue: step-0 x fragments + decay broadcast.
  v16h ax[4];
  float wr8[8];
  {
    const int s0 = dir ? (kS - 1) : 0;
    const int s1 = dir ? (kS - 2) : 1;
    BAR_SIGNAL();                 // open the first "state-ready" round
    prep_step(p, s0, s1, bt, dk, hi, ax, wr8);
  }

  // ------------------------------------------------------------------
  // Persistent recurrence over the sequence (split-barrier pipelined).
  // ------------------------------------------------------------------
  for (int step = 0; step < kS; ++step) {
    const int s = dir ? (kS - 1 - step) : step;

    BAR_WAIT();   // h/c state frags for this step are ready (covered by prep)

    // ---- c_s = tanh(c @ W_d^T + b_d): 8 WMMAs, B/A rotated 1 deep ----
    v8f csacc[2];
    {
      const int cb = kCaOff + (bt * 4) * kFragH + lane * 16;
      const int w0 = kWdbOff + ((j0 + 0) * 4) * kFragH + lane * 16;
      const int w1 = kWdbOff + ((j0 + 4) * 4) * kFragH + lane * 16;
      v16h acf = *(const v16h*)&lds[cb];
      v16h bu0 = *(const v16h*)&lds[w0];
      v16h bu1 = *(const v16h*)&lds[w1];
#pragma unroll
      for (int kf = 0; kf < 4; ++kf) {
        v16h nacf = acf, nbu0 = bu0, nbu1 = bu1;
        if (kf < 3) {
          nacf = *(const v16h*)&lds[cb + (kf + 1) * kFragH];
          nbu0 = *(const v16h*)&lds[w0 + (kf + 1) * kFragH];
          nbu1 = *(const v16h*)&lds[w1 + (kf + 1) * kFragH];
        }
        csacc[0] = (kf == 0) ? wmma16(acf, bu0, vzero) : wmma16(acf, bu0, csacc[0]);
        csacc[1] = (kf == 0) ? wmma16(acf, bu1, vzero) : wmma16(acf, bu1, csacc[1]);
        acf = nacf; bu0 = nbu0; bu1 = nbu1;
      }
    }

    // ---- h A-fragments from LDS ----
    v16h ah[4];
#pragma unroll
    for (int kf = 0; kf < 4; ++kf)
      ah[kf] = *(const v16h*)&lds[kHaOff + (bt * 4 + kf) * kFragH + lane * 16];

    WAIT_DS0();
    BAR_SIGNAL();  // reads done; wait is deferred until just before the stores

    // ---- per column-unit: gates GEMM then immediate pointwise update ----
#pragma unroll
    for (int u = 0; u < 2; ++u) {
      const int j = j0 + u * 4;

      // Two gate pairs, each pair = two independent WMMA chains with
      // B fragments preloaded one kf-slice ahead.
      v8f acc[4];
#pragma unroll
      for (int gp = 0; gp < 2; ++gp) {
        const int g0 = 2 * gp;
        const int f0 = kWcbOff + ((j * 4 + g0) * 8) * kFragH + lane * 16;
        const int f1 = f0 + 8 * kFragH;
        v16h b0 = *(const v16h*)&lds[f0];
        v16h b1 = *(const v16h*)&lds[f1];
        v8f acc0, acc1;
#pragma unroll
        for (int kf = 0; kf < 8; ++kf) {
          v16h nb0 = b0, nb1 = b1;
          if (kf < 7) {
            nb0 = *(const v16h*)&lds[f0 + (kf + 1) * kFragH];
            nb1 = *(const v16h*)&lds[f1 + (kf + 1) * kFragH];
          }
          const v16h a = (kf < 4) ? ax[kf] : ah[kf - 4];
          acc0 = (kf == 0) ? wmma16(a, b0, vzero) : wmma16(a, b0, acc0);
          acc1 = (kf == 0) ? wmma16(a, b1, vzero) : wmma16(a, b1, acc1);
          b0 = nb0; b1 = nb1;
        }
        acc[g0] = acc0;
        acc[g0 + 1] = acc1;
      }

      if (u == 0) BAR_WAIT();  // reads-done round: hidden behind cs + u0 gates

      // pointwise TLSTM cell update + write-back for this unit
      const int col = j * 16 + dk;
      const int kp  = ((j & 1) << 4) + dk;              // k within 32-wide kfrag
      const int tlo = (kp & 8) ? 16 : 0;                // target lane half select
      const int hh  = (kp & 7) | ((kp & 16) >> 1);      // target half index 0..15
      const int fO  = (bt * 4 + (j >> 1)) * kFragH;     // target frag offset
#pragma unroll
      for (int r = 0; r < 8; ++r) {
        const float cshv = tanh_fast(csacc[u][r] + bdv[u]);
        const float cadj = creg[u][r] - cshv + cshv * wr8[r];
        const float ig   = sigm(acc[0][r] + bias[u][0]);
        const float fg   = sigm(acc[1][r] + bias[u][1]);
        const float gg   = tanh_fast(acc[2][r] + bias[u][2]);
        const float og   = sigm(acc[3][r] + bias[u][3]);
        const float cn   = fg * cadj + ig * gg;
        const float hn   = og * tanh_fast(cn);
        creg[u][r] = cn;
        const int slot = (r + 8 * hi + tlo) * 16 + hh;  // scatter into A-frag layout
        lds[kHaOff + fO + slot] = (_Float16)hn;
        lds[kCaOff + fO + slot] = (_Float16)cn;
        p.out[((size_t)s * kB + bt * 16 + r + 8 * hi) * (2 * kH) + dir * kH + col] = hn;
      }
    }

    WAIT_DS0();
    BAR_SIGNAL();  // writes done; completion overlaps the prep below

    // ---- tail: next step's x fragments / decay (independent of LDS state) ----
    {
      const int stepn = (step + 1 < kS) ? (step + 1) : step;
      const int stepp = (step + 2 < kS) ? (step + 2) : (kS - 1);
      const int sn = dir ? (kS - 1 - stepn) : stepn;
      const int sp = dir ? (kS - 1 - stepp) : stepp;
      prep_step(p, sn, sp, bt, dk, hi, ax, wr8);
    }
  }
  BAR_WAIT();  // drain the final writes-done round
}

extern "C" void kernel_launch(void* const* d_in, const int* in_sizes, int n_in,
                              void* d_out, int out_size, void* d_ws, size_t ws_size,
                              hipStream_t stream) {
  (void)in_sizes; (void)n_in; (void)out_size; (void)d_ws; (void)ws_size;
  Params p;
  p.x  = (const float*)d_in[0];
  p.h0 = (const float*)d_in[1];
  p.c0 = (const float*)d_in[2];
  p.dt = (const float*)d_in[3];
  for (int d = 0; d < 2; ++d) {
    const int base = 4 + d * 6;
    p.W_ih[d] = (const float*)d_in[base + 0];
    p.W_hh[d] = (const float*)d_in[base + 1];
    p.b_ih[d] = (const float*)d_in[base + 2];
    p.b_hh[d] = (const float*)d_in[base + 3];
    p.W_d[d]  = (const float*)d_in[base + 4];
    p.b_d[d]  = (const float*)d_in[base + 5];
  }
  p.out = (float*)d_out;

  hipLaunchKernelGGL(tlstm_bidir_kernel, dim3(2), dim3(kThreads), 0, stream, p);
}
